// KernelBRNN_84937273246212
// MI455X (gfx1250) — 8-bit matrix operand order repaired and checked by reading, compiled, not run on hardware
//
#include <hip/hip_runtime.h>

typedef __attribute__((ext_vector_type(8))) int v8i;

#define CARRY   384
#define READD   128
#define DM      512
#define VOC     128
#define NFF     7
#define TT      1024
#define XSTRIDE 528

__global__ void __launch_bounds__(256) cvt_ff_kernel(const float* __restrict__ ff,
                                                     signed char* __restrict__ ff8, int n) {
  int i = blockIdx.x * blockDim.x + threadIdx.x;
  if (i < n) { int R = (i >> 9) & 511, C = i & 511; int src = (i & ~(DM * DM - 1)) + ((R & ~63) + ((R >> 4) & 1) * 8 + ((R >> 5) & 1) * 32 + ((C >> 3) & 1) * 16 + (C & 7)) * DM + (C & ~15) + (R & 15); ff8[i] = (ff[src] >= 0.0f) ? (signed char)1 : (signed char)-1; }
}

__global__ void __launch_bounds__(512) brnn_kernel(
    const int* __restrict__ tokens,
    const float* __restrict__ initial,
    const float* __restrict__ embed,
    const signed char* __restrict__ ffw,
    const float* __restrict__ ffthr,
    const float* __restrict__ head,
    float* __restrict__ out)
{
  __shared__ signed char xbuf[2][16][XSTRIDE];
  __shared__ signed char emb8[VOC][READD];
  __shared__ signed char head8[READD][VOC];
  __shared__ float       thr[NFF][DM];

  const int tid  = threadIdx.x;
  const int wg   = blockIdx.x;
  const int wave = tid >> 5;
  const int lane = tid & 31;
  const int mrow = lane & 15;
  const int half = lane >> 4;

  for (int i = tid; i < VOC * READD; i += 512) {
    ((signed char*)emb8)[i]  = (embed[i] >= 0.f) ? 1 : -1;
    ((signed char*)head8)[i] = (head[((((i >> 7) & ~63) + (((i >> 7) >> 4) & 1) * 8 + (((i >> 7) >> 5) & 1) * 32 + (((i & 127) >> 3) & 1) * 16 + (i & 7)) * VOC) + ((i & 127) & ~15) + ((i >> 7) & 15)]  >= 0.f) ? 1 : -1;
  }
  for (int i = tid; i < NFF * DM; i += 512) ((float*)thr)[i] = ffthr[i];
  for (int i = tid; i < 16 * DM; i += 512)
    xbuf[0][i / DM][i % DM] = (initial[i % DM] >= 0.f) ? 1 : -1;
  __syncthreads();

  int cur = 0;
  for (int t = 0; t < TT; ++t) {
    {
      int row = tid >> 5;
      int co  = (tid & 31) << 2;
      int tok = tokens[(size_t)(wg * 16 + row) * TT + t];
      *(int*)&xbuf[cur][row][CARRY + co] = *(const int*)&emb8[tok][co];
    }
    __syncthreads();

    for (int l = 0; l < NFF; ++l) {
      const int nxt = cur ^ 1;
      v8i acc0 = {}, acc1 = {};
      const int n0 = (wave * 2) * 16;
      const int n1 = n0 + 16;
      const signed char* wl = ffw + (size_t)l * DM * DM;

      for (int kt = 0; kt < 8; ++kt) {
        const signed char* abase = &xbuf[cur][mrow][kt * 64 + half * 8];
        v8i a;
        a[0] = *(const int*)(abase + 0);
        a[1] = *(const int*)(abase + 4);
        a[2] = *(const int*)(abase + 16);
        a[3] = *(const int*)(abase + 20);
        a[4] = *(const int*)(abase + 32);
        a[5] = *(const int*)(abase + 36);
        a[6] = *(const int*)(abase + 48);
        a[7] = *(const int*)(abase + 52);

        const signed char* pf = ffw + (size_t)((l + 1) % NFF) * DM * DM
                              + (size_t)(kt * 64 + lane) * DM + ((wave & 1) ? 256 : 0);
        __builtin_prefetch(pf, 0, 1);

        const signed char* wk = wl + (size_t)(kt * 64 + lane) * DM;
        {
          v8i b;
          const int4 lo = *(const int4*)(wk + n0);
          const int4 hi = *(const int4*)(wk + 32 * DM + n0);
          b[0] = lo.x; b[1] = lo.y; b[2] = lo.z; b[3] = lo.w;
          b[4] = hi.x; b[5] = hi.y; b[6] = hi.z; b[7] = hi.w;
          acc0 = __builtin_amdgcn_wmma_i32_16x16x64_iu8(true, a, true, b, acc0, false, false);
        }
        {
          v8i b;
          const int4 lo = *(const int4*)(wk + n1);
          const int4 hi = *(const int4*)(wk + 32 * DM + n1);
          b[0] = lo.x; b[1] = lo.y; b[2] = lo.z; b[3] = lo.w;
          b[4] = hi.x; b[5] = hi.y; b[6] = hi.z; b[7] = hi.w;
          acc1 = __builtin_amdgcn_wmma_i32_16x16x64_iu8(true, a, true, b, acc1, false, false);
        }
      }

      {
        const int ngA = n0 + mrow, ngB = n1 + mrow;
        const float tA = thr[l][ngA], tB = thr[l][ngB];
        #pragma unroll
        for (int r = 0; r < 8; ++r) {
          int m = half * 8 + r;
          xbuf[nxt][m][ngA] = ((float)acc0[r] >= tA) ? 1 : -1;
          xbuf[nxt][m][ngB] = ((float)acc1[r] >= tB) ? 1 : -1;
        }
      }
      __syncthreads();
      cur = nxt;
    }

    if (wave < 8) {
      v8i acc = {};
      const int nG = wave * 16;
      for (int kt = 0; kt < 2; ++kt) {
        const signed char* abase = &xbuf[cur][mrow][CARRY + kt * 64 + half * 8];
        v8i a;
        a[0] = *(const int*)(abase + 0);
        a[1] = *(const int*)(abase + 4);
        a[2] = *(const int*)(abase + 16);
        a[3] = *(const int*)(abase + 20);
        a[4] = *(const int*)(abase + 32);
        a[5] = *(const int*)(abase + 36);
        a[6] = *(const int*)(abase + 48);
        a[7] = *(const int*)(abase + 52);
        v8i b;
        const signed char* hk = &head8[kt * 64 + lane][0];
        const int4 lo = *(const int4*)(hk + nG);
        const int4 hi = *(const int4*)(hk + 32 * VOC + nG);
        b[0] = lo.x; b[1] = lo.y; b[2] = lo.z; b[3] = lo.w;
        b[4] = hi.x; b[5] = hi.y; b[6] = hi.z; b[7] = hi.w;
        acc = __builtin_amdgcn_wmma_i32_16x16x64_iu8(true, a, true, b, acc, false, false);
      }
      const int v = nG + mrow;
      #pragma unroll
      for (int r = 0; r < 8; ++r) {
        int m = half * 8 + r;
        out[((size_t)(wg * 16 + m) * TT + t) * VOC + v] = (float)acc[r];
      }
    }
    __syncthreads();
  }
}

extern "C" void kernel_launch(void* const* d_in, const int* in_sizes, int n_in,
                              void* d_out, int out_size, void* d_ws, size_t ws_size,
                              hipStream_t stream) {
  const int*   tokens  = (const int*)d_in[0];
  const float* initial = (const float*)d_in[1];
  const float* embed   = (const float*)d_in[2];
  const float* ff      = (const float*)d_in[3];
  const float* ffthr   = (const float*)d_in[4];
  const float* head    = (const float*)d_in[5];
  float* out = (float*)d_out;

  signed char* ff8 = (signed char*)d_ws;
  const int n = NFF * DM * DM;
  cvt_ff_kernel<<<(n + 255) / 256, 256, 0, stream>>>(ff, ff8, n);

  brnn_kernel<<<8, 512, 0, stream>>>(tokens, initial, embed, ff8, ffthr, head, out);
}
